// GCN_40888088658390
// MI455X (gfx1250) — matrix operand order repaired and checked by reading, not compiled, not run on hardware
//
#include <hip/hip_runtime.h>

// ---------------- problem constants (D/L/OUT fixed by the model) -----------
#define DD     128
#define LAYERS 4
#define OUTD   10
#define BN_EPS 1e-5f

typedef __attribute__((ext_vector_type(16))) __bf16 v16bf;
typedef __attribute__((ext_vector_type(8)))  __bf16 v8bf;
typedef __attribute__((ext_vector_type(8)))  float  v8f;

// ---------------------------------------------------------------------------
// generic zero fill (grid-stride)
__global__ void gcn_zero_f32(float* __restrict__ p, long n) {
    long i = (long)blockIdx.x * blockDim.x + threadIdx.x;
    long stride = (long)gridDim.x * blockDim.x;
    for (; i < n; i += stride) p[i] = 0.0f;
}

// degree histogram (int atomics); deg buffers are reused as norm buffers
__global__ void gcn_degree(const int* __restrict__ src, const int* __restrict__ dst,
                           int* __restrict__ deg_out, int* __restrict__ deg_in, int E) {
    int e = blockIdx.x * blockDim.x + threadIdx.x;
    if (e >= E) return;
    atomicAdd(&deg_out[src[e]], 1);
    atomicAdd(&deg_in[dst[e]], 1);
}

// in-place int degree -> float deg^-1/2 with clamp to >=1
__global__ void gcn_norm(float* __restrict__ buf, long n) {
    long i = (long)blockIdx.x * blockDim.x + threadIdx.x;
    if (i >= n) return;
    int dg = ((const int*)buf)[i];
    float d = dg < 1 ? 1.0f : (float)dg;
    buf[i] = rsqrtf(d);
}

// convert all GraphConv weights f32 -> bf16 once per call
__global__ void gcn_cvt_w(const float* __restrict__ w, __bf16* __restrict__ wbf, long n) {
    long i = (long)blockIdx.x * blockDim.x + threadIdx.x;
    if (i >= n) return;
    wbf[i] = (__bf16)w[i];
}

// SpMM scatter: agg[dst] += h[src] * norm_src[src].  One wave per edge,
// each lane owns a float4 feature chunk (L2-resident atomics).
__global__ void gcn_scatter(const float* __restrict__ hin,
                            const float* __restrict__ norm_src,
                            const int* __restrict__ src, const int* __restrict__ dst,
                            float* __restrict__ agg, int E) {
    int t = blockIdx.x * blockDim.x + threadIdx.x;
    int e = t >> 5;
    int lane = t & 31;
    if (e >= E) return;
    int s = src[e], d = dst[e];
    float ns = norm_src[s];
    float4 v = ((const float4*)(hin + (size_t)s * DD))[lane];
    float* p = agg + (size_t)d * DD + lane * 4;
    atomicAdd(p + 0, v.x * ns);
    atomicAdd(p + 1, v.y * ns);
    atomicAdd(p + 2, v.z * ns);
    atomicAdd(p + 3, v.w * ns);
}

// ---------------------------------------------------------------------------
// Fused:  Y = (diag(norm_dst) * AGG) @ W + b   (bf16 WMMA, f32 accumulate)
//         plus per-feature sum / sum^2 partials for BatchNorm.
// 256 threads = 8 waves; block computes 128 rows x 128 cols; in-place Y==AGG.
__global__ __launch_bounds__(256)
void gcn_gemm_wmma(const float* __restrict__ agg,
                   const float* __restrict__ norm_dst,
                   const __bf16* __restrict__ Wbf,   // 128x128 layer slice
                   const float* __restrict__ bias,   // 128
                   float* __restrict__ y,            // == agg (in-place, block-private rows)
                   float* __restrict__ stats,        // [256]: sum[128], sumsq[128]
                   int n_rows) {
    __shared__ __bf16 sA[128][DD + 8];
    __shared__ __bf16 sB[128][DD + 8];
    __shared__ float s_sum[DD];
    __shared__ float s_sq[DD];

    const int tid = threadIdx.x;
    const int r0  = blockIdx.x * 128;

    if (tid < DD) { s_sum[tid] = 0.0f; s_sq[tid] = 0.0f; }

    // stage A: row-scale by norm_dst, convert f32 -> bf16
#pragma unroll
    for (int j = 0; j < 16; ++j) {
        int idx4 = tid + j * 256;            // 0..4095 float4 slots
        int row  = idx4 >> 5;                // 32 float4 per row
        int c4   = idx4 & 31;
        int grow = r0 + row;
        float4 v = make_float4(0.f, 0.f, 0.f, 0.f);
        float  sc = 0.0f;
        if (grow < n_rows) {
            v  = ((const float4*)(agg + (size_t)grow * DD))[c4];
            sc = norm_dst[grow];
        }
        __bf16* d = &sA[row][c4 * 4];
        d[0] = (__bf16)(v.x * sc); d[1] = (__bf16)(v.y * sc);
        d[2] = (__bf16)(v.z * sc); d[3] = (__bf16)(v.w * sc);
    }
    // stage B (already bf16)
#pragma unroll
    for (int j = 0; j < 8; ++j) {
        int idx = tid + j * 256;             // 0..2047 8-element chunks
        int row = idx >> 4;
        int c8  = idx & 15;
        *(v8bf*)&sB[row][c8 * 8] = *(const v8bf*)(Wbf + row * DD + c8 * 8);
    }
    __syncthreads();

    const int wave = tid >> 5;
    const int lane = tid & 31;
    const int hi   = lane >> 4;              // lane group 0/1
    const int lmod = lane & 15;
    const int mrow = wave * 16;              // wave's 16-row tile in block

    v8f acc[8] = {};

#pragma unroll
    for (int kk = 0; kk < 4; ++kk) {
        const int kbase = kk * 32;
        // A fragment: lane m -> row m, K {0..7,16..23}; lane m+16 -> K {8..15,24..31}
        v8bf alo = *(const v8bf*)&sA[mrow + lmod][kbase + hi * 8];
        v8bf ahi = *(const v8bf*)&sA[mrow + lmod][kbase + 16 + hi * 8];
        v16bf afrag;
#pragma unroll
        for (int q = 0; q < 8; ++q) { afrag[q] = alo[q]; afrag[8 + q] = ahi[q]; }

#pragma unroll
        for (int nt = 0; nt < 8; ++nt) {
            // B fragment: lane n -> col N = nt*16 + n, K {0..7,16..23}; lane n+16 -> K {8..15,24..31}
            v16bf bfrag;
#pragma unroll
            for (int q = 0; q < 8; ++q) { bfrag[q] = sB[kbase + hi * 8 + q][nt * 16 + lmod]; bfrag[8 + q] = sB[kbase + 16 + hi * 8 + q][nt * 16 + lmod]; }

            acc[nt] = __builtin_amdgcn_wmma_f32_16x16x32_bf16(
                false, afrag, false, bfrag, (short)0, acc[nt], false, false);
        }
    }

    // C layout: VGPR r -> row r (lanes 0-15) / 8+r (lanes 16-31); col = nt*16+lmod
#pragma unroll
    for (int nt = 0; nt < 8; ++nt) {
        const int col = nt * 16 + lmod;
        const float bv = bias[col];
        float s = 0.0f, sq = 0.0f;
#pragma unroll
        for (int r = 0; r < 8; ++r) {
            int grow = r0 + mrow + hi * 8 + r;
            if (grow < n_rows) {
                float v = acc[nt][r] + bv;
                y[(size_t)grow * DD + col] = v;
                s += v; sq += v * v;
            }
        }
        atomicAdd(&s_sum[col], s);
        atomicAdd(&s_sq[col], sq);
    }
    __syncthreads();
    if (tid < DD) {
        atomicAdd(&stats[tid], s_sum[tid]);
        atomicAdd(&stats[DD + tid], s_sq[tid]);
    }
}

// BatchNorm (batch stats) + ReLU + optional residual; in-place on hcur
__global__ void gcn_bn_apply(const float* __restrict__ y, float* __restrict__ hcur,
                             const float* __restrict__ stats,
                             const float* __restrict__ gamma, const float* __restrict__ beta,
                             int with_res, int n_rows) {
    long i = (long)blockIdx.x * blockDim.x + threadIdx.x;
    long total = (long)n_rows * DD;
    if (i >= total) return;
    int col = (int)(i & (DD - 1));
    float invN = 1.0f / (float)n_rows;
    float mu   = stats[col] * invN;
    float var  = stats[DD + col] * invN - mu * mu;
    float rs   = rsqrtf(var + BN_EPS);
    float v = (y[i] - mu) * rs * gamma[col] + beta[col];
    v = fmaxf(v, 0.0f);
    hcur[i] = with_res ? (v + hcur[i]) : v;
}

// SumPooling: pooled[gid] += h[row]; wave per row, float4 per lane
__global__ void gcn_pool(const float* __restrict__ h, const int* __restrict__ ngid,
                         float* __restrict__ pooled, int n_rows) {
    int t = blockIdx.x * blockDim.x + threadIdx.x;
    int row = t >> 5, lane = t & 31;
    if (row >= n_rows) return;
    int g = ngid[row];
    float4 v = ((const float4*)(h + (size_t)row * DD))[lane];
    float* p = pooled + (size_t)g * DD + lane * 4;
    atomicAdd(p + 0, v.x);
    atomicAdd(p + 1, v.y);
    atomicAdd(p + 2, v.z);
    atomicAdd(p + 3, v.w);
}

// prediction head: [G,128] @ [128,10] + bp  (tiny, one thread per output)
__global__ void gcn_head(const float* __restrict__ pooled, const float* __restrict__ Wp,
                         const float* __restrict__ bp, float* __restrict__ out, int G) {
    int t = blockIdx.x * blockDim.x + threadIdx.x;
    if (t >= G * OUTD) return;
    int g = t / OUTD, o = t - g * OUTD;
    float a = bp[o];
    const float* pr = pooled + (size_t)g * DD;
#pragma unroll 16
    for (int k = 0; k < DD; ++k) a += pr[k] * Wp[k * OUTD + o];
    out[t] = a;
}

// ---------------------------------------------------------------------------
static inline int zgrid(long n) {
    long b = (n + 255) / 256;
    return (int)(b < 16384 ? b : 16384);
}

extern "C" void kernel_launch(void* const* d_in, const int* in_sizes, int n_in,
                              void* d_out, int out_size, void* d_ws, size_t ws_size,
                              hipStream_t stream) {
    const float* h_in  = (const float*)d_in[0];   // [N,128]
    const float* W     = (const float*)d_in[1];   // [4,128,128]
    const float* b     = (const float*)d_in[2];   // [4,128]
    const float* gamma = (const float*)d_in[3];   // [4,128]
    const float* beta  = (const float*)d_in[4];   // [4,128]
    const float* Wp    = (const float*)d_in[5];   // [128,10]
    const float* bp    = (const float*)d_in[6];   // [10]
    const int*   src   = (const int*)d_in[7];     // [E]
    const int*   dst   = (const int*)d_in[8];     // [E]
    const int*   ngid  = (const int*)d_in[9];     // [N]

    const int N = in_sizes[0] / DD;
    const int E = in_sizes[7];
    const int G = out_size / OUTD;

    // workspace carve-up (f32 units)
    float*  agg    = (float*)d_ws;                 // N*128
    float*  stats  = agg + (size_t)N * DD;         // 256 (contiguous -> zeroed with agg)
    float*  hcur   = stats + 256;                  // N*128
    float*  nsrc   = hcur + (size_t)N * DD;        // N  (int degrees then f32 norms)
    float*  ndst   = nsrc + N;                     // N
    float*  pooled = ndst + N;                     // G*128
    __bf16* Wbf    = (__bf16*)(pooled + (size_t)G * DD);  // 4*128*128 bf16

    // degrees -> norms
    gcn_zero_f32<<<zgrid(2L * N), 256, 0, stream>>>(nsrc, 2L * N);
    gcn_degree<<<(E + 255) / 256, 256, 0, stream>>>(src, dst, (int*)nsrc, (int*)ndst, E);
    gcn_norm<<<(2L * N + 255) / 256, 256, 0, stream>>>(nsrc, 2L * N);

    // weights -> bf16
    const long wElems = (long)LAYERS * DD * DD;
    gcn_cvt_w<<<(wElems + 255) / 256, 256, 0, stream>>>(W, Wbf, wElems);

    const int gemmGrid = (N + 127) / 128;
    for (int i = 0; i < LAYERS; ++i) {
        gcn_zero_f32<<<zgrid((long)N * DD + 256), 256, 0, stream>>>(agg, (long)N * DD + 256);
        gcn_scatter<<<(int)(((long)E * 32 + 255) / 256), 256, 0, stream>>>(
            i == 0 ? h_in : hcur, nsrc, src, dst, agg, E);
        gcn_gemm_wmma<<<gemmGrid, 256, 0, stream>>>(
            agg, ndst, Wbf + (size_t)i * DD * DD, b + i * DD, agg, stats, N);
        gcn_bn_apply<<<(int)(((long)N * DD + 255) / 256), 256, 0, stream>>>(
            agg, hcur, stats, gamma + i * DD, beta + i * DD, i != 0, N);
    }

    // pooling + head
    gcn_zero_f32<<<zgrid((long)G * DD), 256, 0, stream>>>(pooled, (long)G * DD);
    gcn_pool<<<(int)(((long)N * 32 + 255) / 256), 256, 0, stream>>>(hcur, ngid, pooled, N);
    gcn_head<<<(G * OUTD + 255) / 256, 256, 0, stream>>>(pooled, Wp, bp, (float*)d_out, G);
}
